// SelfAttentionHead_10153302688075
// MI455X (gfx1250) — compile-verified, matrix operand order checked by reading, not run on hardware
//
#include <hip/hip_runtime.h>
#include <hip/hip_bf16.h>

// Problem constants (B, S, D, H) from the reference.
constexpr int Bc = 4;
constexpr int Sc = 4096;
constexpr int Dc = 1024;
constexpr int Hc = 128;

typedef float          v8f    __attribute__((ext_vector_type(8)));
typedef __bf16         v16bf  __attribute__((ext_vector_type(16)));
typedef unsigned short u16x8  __attribute__((ext_vector_type(8)));
typedef unsigned short u16x16 __attribute__((ext_vector_type(16)));

// fp32 -> bf16 (round-to-nearest-even) via HIP's intrinsic path.
__device__ __forceinline__ unsigned short f2bf(float f) {
  __hip_bfloat16 h = __float2bfloat16(f);
  return __builtin_bit_cast(unsigned short, h);
}

// Load 16 bf16 fragment elements as two contiguous 16B chunks (p, p+16 shorts).
// Matches the CDNA5 16-bit A/B fragment layout: per lane, elements 0..7 are
// K = kbase+0..7 and elements 8..15 are K = kbase+16..23 (kbase = (lane>=16)?8:0).
__device__ __forceinline__ v16bf frag16(const unsigned short* p) {
  u16x8 lo = __builtin_bit_cast(u16x8, *(const uint4*)(p));
  u16x8 hi = __builtin_bit_cast(u16x8, *(const uint4*)(p + 16));
  u16x16 c = __builtin_shufflevector(lo, hi, 0,1,2,3,4,5,6,7,8,9,10,11,12,13,14,15);
  return __builtin_bit_cast(v16bf, c);
}

__device__ __forceinline__ v8f wmma_bf16(v16bf a, v16bf b, v8f c) {
  // (neg_a, A, neg_b, B, c_mod, C, reuse_a, reuse_b)
  return __builtin_amdgcn_wmma_f32_16x16x32_bf16(false, a, false, b, (short)0, c,
                                                 false, false);
}

// ASYNCcnt-tracked global -> LDS DMA (16B per lane). vdst = LDS byte address,
// vaddr = per-lane 32-bit byte offset, saddr = uniform 64-bit base (GVS mode).
__device__ __forceinline__ void async_load_b128(unsigned lds_addr, unsigned goff,
                                                const void* gbase) {
  asm volatile("global_load_async_to_lds_b128 %0, %1, %2 offset:0"
               :: "v"(lds_addr), "v"(goff), "s"(gbase)
               : "memory");
}

// ---------------------------------------------------------------------------
// Pass 0: bulk f32 -> bf16 conversion (memory-bound pre-pass; keeps ALL
// conversion VALU work out of the matmul inner loops).
// ---------------------------------------------------------------------------
__global__ __launch_bounds__(256) void cvt_bf16_kernel(
    const float* __restrict__ src, unsigned short* __restrict__ dst, int n4) {
  const int i = blockIdx.x * 256 + threadIdx.x;
  if (i >= n4) return;
  const float4 v = ((const float4*)src)[i];
  ushort4 o;
  o.x = f2bf(v.x);
  o.y = f2bf(v.y);
  o.z = f2bf(v.z);
  o.w = f2bf(v.w);
  ((ushort4*)dst)[i] = o;
}

// ---------------------------------------------------------------------------
// Pass 1: fused Q/K/V projection. One wave owns a 16-row s-tile and computes
// the full 16x384 output slab (8 h-subtiles x {Q,K,V}). The per-K-step weight
// slice (384 rows x 32 k = 24KB bf16) is staged into LDS once per BLOCK with
// double-buffered global_load_async_to_lds_b128 (ASYNCcnt), overlapping the
// next stage's DMA with the current stage's 24 WMMAs. B-fragments then come
// from LDS (ds_load_b128, 4x less L2 traffic); A-fragments stream from
// global. Q gets 1/sqrt(H) folded in; V is stored TRANSPOSED (B,H,S) so the
// attention pass reads P*V B-fragments as contiguous 16B loads out of L2.
// ---------------------------------------------------------------------------
__global__ __launch_bounds__(128) void qkv_proj_kernel(
    const unsigned short* __restrict__ Xb,   // (B,S,D) bf16
    const unsigned short* __restrict__ Wb,   // (3,H,D) bf16: Wq,Wk,Wv
    unsigned short* __restrict__ Qb,
    unsigned short* __restrict__ Kb,
    unsigned short* __restrict__ Vt) {
  // Double-buffered weight slice: [buf][row=3*128][k=32] bf16, 24KB per buffer.
  __shared__ unsigned short ldsW[2][384 * 32];

  const int b    = blockIdx.y;
  const int tid  = threadIdx.x;
  const int wave = tid >> 5;
  const int lane = tid & 31;
  const int l15  = lane & 15;
  const int half = lane >> 4;
  const int kb8  = half * 8;                 // fragment K base per half-wave

  const int sTile = blockIdx.x * 4 + wave;
  const int sBase = sTile * 16;

  const unsigned short* xrow = Xb + ((size_t)b * Sc + sBase + l15) * Dc;

  v8f acc[3][8];
#pragma unroll
  for (int w = 0; w < 3; ++w)
#pragma unroll
    for (int j = 0; j < 8; ++j)
      acc[w][j] = (v8f){0.f, 0.f, 0.f, 0.f, 0.f, 0.f, 0.f, 0.f};

  // Issue one stage of W DMA: 1536 x 16B transfers, 12 async ops per thread.
  // transfer t: row = t>>2 (0..383), chunk = t&3; 64B per row per stage.
  auto issue_stage = [&](int k0, int buf) {
    const unsigned lds_base = (unsigned)(uintptr_t)(&ldsW[buf][0]);
#pragma unroll
    for (int i = 0; i < 12; ++i) {
      const int t    = i * 128 + tid;
      const int row  = t >> 2;
      const int chk  = t & 3;
      const unsigned goff = (unsigned)(row * Dc + k0 + chk * 8) * 2u;
      const unsigned loff = lds_base + (unsigned)(row * 64 + chk * 16);
      async_load_b128(loff, goff, Wb);
    }
  };

  issue_stage(0, 0);

  for (int step = 0; step < Dc / 32; ++step) {
    const int k0  = step * 32;
    const int cur = step & 1;

    if (step + 1 < Dc / 32) {
      // Previous compute finished block-wide (barrier at loop tail), so the
      // other buffer is free: start its DMA now, then wait only for `cur`.
      issue_stage(k0 + 32, cur ^ 1);
      asm volatile("s_wait_asynccnt 12" ::: "memory");  // cur landed, next in flight
    } else {
      asm volatile("s_wait_asynccnt 0" ::: "memory");
    }
    __syncthreads();                       // cur visible to all 4 waves

    const v16bf af = frag16(xrow + k0 + kb8);
#pragma unroll
    for (int w = 0; w < 3; ++w) {
#pragma unroll
      for (int j = 0; j < 8; ++j) {
        const unsigned short* wp = &ldsW[cur][(w * Hc + j * 16 + l15) * 32 + kb8];
        acc[w][j] = wmma_bf16(af, frag16(wp), acc[w][j]);
      }
    }
    __syncthreads();                       // all waves done reading cur
  }

  // C layout: element r -> row M = r + 8*half, col N = lane&15
  const float scaleQ = 0.08838834764831843f;  // 1/sqrt(128)
#pragma unroll
  for (int j = 0; j < 8; ++j) {
    const int h = j * 16 + l15;
#pragma unroll
    for (int r = 0; r < 8; ++r) {
      const int s = sBase + r + half * 8;
      Qb[((size_t)b * Sc + s) * Hc + h] = f2bf(acc[0][j][r] * scaleQ);
      Kb[((size_t)b * Sc + s) * Hc + h] = f2bf(acc[1][j][r]);
      Vt[((size_t)b * Hc + h) * Sc + s] = f2bf(acc[2][j][r]);  // transposed V
    }
  }
}

// ---------------------------------------------------------------------------
// Pass 2: causal flash attention. One wave owns a 16-row query tile, streams
// key blocks of 32 (two 16x16 score accumulators). Q@K^T: 8 WMMAs/block over
// H=128. Online softmax with half-wave shfl_xor reductions (score rows live
// in one VGPR slot across a 16-lane half). P converted to an A-fragment via a
// 1KB/wave LDS bounce (s_wait_dscnt for intra-wave ordering), then P@V: 8
// WMMAs/block into the 16x128 f32 O accumulator, whose layout matches the
// score layout so alpha rescaling is a per-element multiply. No block-wide
// barriers: waves have causally different trip counts.
// ---------------------------------------------------------------------------
__global__ __launch_bounds__(128) void attn_kernel(
    const unsigned short* __restrict__ Qb,
    const unsigned short* __restrict__ Kb,
    const unsigned short* __restrict__ Vt,
    float* __restrict__ out) {
  __shared__ unsigned short ldsP[4][16 * 32];

  const int b    = blockIdx.y;
  const int wave = threadIdx.x >> 5;
  const int lane = threadIdx.x & 31;
  const int l15  = lane & 15;
  const int half = lane >> 4;
  const int kb8  = half * 8;

  const int qTile = blockIdx.x * 4 + wave;
  const int qBase = qTile * 16;

  // Persistent Q fragments: 16x128 bf16 as 4 A-fragments (1/sqrt(H) pre-folded)
  v16bf qf[4];
  {
    const unsigned short* qrow = Qb + ((size_t)b * Sc + qBase + l15) * Hc;
#pragma unroll
    for (int f = 0; f < 4; ++f) qf[f] = frag16(qrow + f * 32 + kb8);
  }

  v8f O[8];
#pragma unroll
  for (int j = 0; j < 8; ++j) O[j] = (v8f){0.f, 0.f, 0.f, 0.f, 0.f, 0.f, 0.f, 0.f};
  float mrow[8], lrow[8];
#pragma unroll
  for (int r = 0; r < 8; ++r) { mrow[r] = -__builtin_inff(); lrow[r] = 0.f; }

  const unsigned short* kbasep = Kb + (size_t)b * Sc * Hc;
  const unsigned short* vbasep = Vt + (size_t)b * Hc * Sc;

  const int nkb = (qTile >> 1) + 1;          // key blocks of 32 up to diagonal
  for (int kblk = 0; kblk < nkb; ++kblk) {
    const int keyBase = kblk * 32;

    // ---- S = Q @ K^T (scores for 16x32 keys) ----
    const unsigned short* krow0 = kbasep + (size_t)(keyBase + l15) * Hc;
    const unsigned short* krow1 = krow0 + 16 * Hc;
    if (kblk + 1 < nkb) {                    // prefetch next K block
      __builtin_prefetch(krow0 + 32 * Hc, 0, 3);
      __builtin_prefetch(krow1 + 32 * Hc, 0, 3);
    }
    v8f s0 = (v8f){0.f, 0.f, 0.f, 0.f, 0.f, 0.f, 0.f, 0.f};
    v8f s1 = (v8f){0.f, 0.f, 0.f, 0.f, 0.f, 0.f, 0.f, 0.f};
#pragma unroll
    for (int f = 0; f < 4; ++f) {
      v16bf kf0 = frag16(krow0 + f * 32 + kb8);
      v16bf kf1 = frag16(krow1 + f * 32 + kb8);
      s0 = wmma_bf16(qf[f], kf0, s0);
      s1 = wmma_bf16(qf[f], kf1, s1);
    }

    // ---- causal mask (only blocks touching the diagonal) ----
    if (keyBase + 31 > qBase) {
#pragma unroll
      for (int r = 0; r < 8; ++r) {
        const int row = qBase + r + half * 8;
        if (keyBase + l15 > row)      s0[r] = -__builtin_inff();
        if (keyBase + 16 + l15 > row) s1[r] = -__builtin_inff();
      }
    }

    // ---- online softmax; write P (bf16) into LDS in C-layout ----
#pragma unroll
    for (int r = 0; r < 8; ++r) {
      float mx = fmaxf(s0[r], s1[r]);
      mx = fmaxf(mx, __shfl_xor(mx, 1, 32));
      mx = fmaxf(mx, __shfl_xor(mx, 2, 32));
      mx = fmaxf(mx, __shfl_xor(mx, 4, 32));
      mx = fmaxf(mx, __shfl_xor(mx, 8, 32));
      const float mnew  = fmaxf(mrow[r], mx);
      const float alpha = __expf(mrow[r] - mnew);
      mrow[r] = mnew;
      const float p0 = __expf(s0[r] - mnew);
      const float p1 = __expf(s1[r] - mnew);
      float ps = p0 + p1;
      ps += __shfl_xor(ps, 1, 32);
      ps += __shfl_xor(ps, 2, 32);
      ps += __shfl_xor(ps, 4, 32);
      ps += __shfl_xor(ps, 8, 32);
      lrow[r] = lrow[r] * alpha + ps;
#pragma unroll
      for (int j = 0; j < 8; ++j) O[j][r] *= alpha;   // C-layout matches scores
      const int crow = (r + half * 8) * 32;
      ldsP[wave][crow + l15]      = f2bf(p0);
      ldsP[wave][crow + 16 + l15] = f2bf(p1);
    }

    // Intra-wave LDS ordering: C-layout stores -> A-layout loads.
    asm volatile("s_wait_dscnt 0" ::: "memory");

    v16bf pf = frag16(&ldsP[wave][l15 * 32 + kb8]);

    // ---- O += P @ V  (V^T layout -> contiguous 16B B-fragment loads) ----
#pragma unroll
    for (int j = 0; j < 8; ++j) {
      const unsigned short* vp =
          vbasep + (size_t)(j * 16 + l15) * Sc + keyBase + kb8;
      v16bf vf = frag16(vp);
      O[j] = wmma_bf16(pf, vf, O[j]);
    }
    asm volatile("s_wait_dscnt 0" ::: "memory");  // WAR: next iter rewrites ldsP
  }

  // ---- epilogue: out = O / l ----
#pragma unroll
  for (int r = 0; r < 8; ++r) {
    const float inv = 1.0f / lrow[r];
    const int row   = qBase + r + half * 8;
#pragma unroll
    for (int j = 0; j < 8; ++j) {
      out[((size_t)b * Sc + row) * Hc + j * 16 + l15] = O[j][r] * inv;
    }
  }
}

extern "C" void kernel_launch(void* const* d_in, const int* in_sizes, int n_in,
                              void* d_out, int out_size, void* d_ws, size_t ws_size,
                              hipStream_t stream) {
  const float* x  = (const float*)d_in[0];
  const float* Wq = (const float*)d_in[1];
  const float* Wk = (const float*)d_in[2];
  const float* Wv = (const float*)d_in[3];

  const size_t xElems = (size_t)Bc * Sc * Dc;      // 16.8M
  const size_t wElems = (size_t)Hc * Dc;           // 131072 per matrix
  const size_t pElems = (size_t)Bc * Sc * Hc;      // 2.1M per tensor

  unsigned short* Xb = (unsigned short*)d_ws;      // bf16 x       (B,S,D)  32MB
  unsigned short* Wb = Xb + xElems;                // bf16 W       (3,H,D)  768KB
  unsigned short* Qb = Wb + 3 * wElems;            // bf16 Q*scale (B,S,H)  4MB
  unsigned short* Kb = Qb + pElems;                // bf16 K       (B,S,H)  4MB
  unsigned short* Vt = Kb + pElems;                // bf16 V^T     (B,H,S)  4MB

  // Pass 0: bulk conversions (memory-bound, ~4us of HBM traffic total).
  cvt_bf16_kernel<<<(int)(xElems / 4 + 255) / 256, 256, 0, stream>>>(x, Xb,
                                                                     (int)(xElems / 4));
  cvt_bf16_kernel<<<(int)(wElems / 4 + 255) / 256, 256, 0, stream>>>(
      Wq, Wb + 0 * wElems, (int)(wElems / 4));
  cvt_bf16_kernel<<<(int)(wElems / 4 + 255) / 256, 256, 0, stream>>>(
      Wk, Wb + 1 * wElems, (int)(wElems / 4));
  cvt_bf16_kernel<<<(int)(wElems / 4 + 255) / 256, 256, 0, stream>>>(
      Wv, Wb + 2 * wElems, (int)(wElems / 4));

  // Pass 1: fused QKV projection (24 WMMAs per K-step per wave, async-staged W).
  dim3 gp(Sc / (16 * 4), Bc);
  qkv_proj_kernel<<<gp, 128, 0, stream>>>(Xb, Wb, Qb, Kb, Vt);

  // Pass 2: causal flash attention.
  dim3 ga(Sc / (16 * 4), Bc);
  attn_kernel<<<ga, 128, 0, stream>>>(Qb, Kb, Vt, (float*)d_out);
}
